// Uncond_MTPN_EncoderLayer_72928544686649
// MI455X (gfx1250) — matrix operand order repaired and checked by reading, not compiled, not run on hardware
//
#include <hip/hip_runtime.h>

#define B_  4
#define S_  1024
#define D_  1024
#define FD_ 4096
#define MM_ 512
#define H_  16
#define HD_ 64
#define L_  2
#define N_  (B_ * S_)

typedef __bf16 bf16;
typedef __bf16 bf16x16 __attribute__((ext_vector_type(16)));
typedef float  floatx8 __attribute__((ext_vector_type(8)));
typedef float  vfloat4 __attribute__((ext_vector_type(4)));

__device__ __forceinline__ bf16 f2bf(float f) {
  unsigned u = __float_as_uint(f);
  u += 0x7FFFu + ((u >> 16) & 1u);           // round-to-nearest-even
  unsigned short h = (unsigned short)(u >> 16);
  bf16 r;
  __builtin_memcpy(&r, &h, 2);
  return r;
}

__device__ __forceinline__ floatx8 wmma_bf(bf16x16 a, bf16x16 b, floatx8 c) {
  return __builtin_amdgcn_wmma_f32_16x16x32_bf16(false, a, false, b, (short)0, c,
                                                 false, false);
}

__device__ __forceinline__ void cp16(void* dst, const void* src) {
  *reinterpret_cast<vfloat4*>(dst) = *reinterpret_cast<const vfloat4*>(src);
}

// ---------------------------------------------------------------------------
// GEMM: C(MxN,f32) = A(MxK,bf16 row-major) * B + bias, with B given
// PRE-TRANSPOSED as Bt (N x K, bf16 row-major).  Tile 128x128 per 256-thread
// block; 8 waves as 4x2, each wave 32x64 = 2x4 WMMA tiles.  Double-buffered
// LDS staging: next tile's global loads are issued before the WMMA burst.
// Requires M%128==0, N%128==0, K%32==0 (true for every GEMM in this net).
// ---------------------------------------------------------------------------
#define GBM 128
#define GBN 128
#define GBK 32
#define LAS (GBK + 8)   // LDS stride (elems); keeps 16B alignment of fragments
#define LBS (GBK + 8)

__global__ __launch_bounds__(256)
void k_gemm_bf16(const bf16* __restrict__ A, const bf16* __restrict__ Bt,
                 const float* __restrict__ bias,
                 float* __restrict__ outF, bf16* __restrict__ outB,
                 int Mr, int Nc, int Kd, int relu)
{
  __shared__ __align__(16) bf16 lA[2][GBM * LAS];
  __shared__ __align__(16) bf16 lB[2][GBN * LBS];   // [n][k]

  const int tid  = threadIdx.x;
  const int lane = tid & 31;
  const int wave = tid >> 5;          // 0..7
  const int wm   = wave >> 1;         // 0..3  (row group: 32 rows each)
  const int wn   = wave & 1;          // 0..1  (col group: 64 cols each)
  const int half = lane >> 4;
  const int l16  = lane & 15;
  const int bm0  = blockIdx.y * GBM;
  const int bn0  = blockIdx.x * GBN;

  const int srow = tid >> 1;          // 0..127: staging row
  const int skc  = (tid & 1) * 16;    // 0 or 16: staging k-chunk
  const int sa   = srow * LAS + skc;
  const int sb   = srow * LBS + skc;

  const bf16* aRow = &A[(size_t)(bm0 + srow) * Kd + skc];
  const bf16* bRow = &Bt[(size_t)(bn0 + srow) * Kd + skc];

  // preload tile 0
  vfloat4 ra0 = *(const vfloat4*)(aRow);
  vfloat4 ra1 = *(const vfloat4*)(aRow + 8);
  vfloat4 rb0 = *(const vfloat4*)(bRow);
  vfloat4 rb1 = *(const vfloat4*)(bRow + 8);
  *(vfloat4*)&lA[0][sa]     = ra0;
  *(vfloat4*)&lA[0][sa + 8] = ra1;
  *(vfloat4*)&lB[0][sb]     = rb0;
  *(vfloat4*)&lB[0][sb + 8] = rb1;

  floatx8 acc[2][4] = {};
  int buf = 0;

  for (int k0 = 0; k0 < Kd; k0 += GBK) {
    const bool hasnext = (k0 + GBK) < Kd;
    if (hasnext) {
      // issue next tile's loads now; latency overlaps the WMMA burst below
      const bf16* an = aRow + k0 + GBK;
      const bf16* bn = bRow + k0 + GBK;
      ra0 = *(const vfloat4*)(an);
      ra1 = *(const vfloat4*)(an + 8);
      rb0 = *(const vfloat4*)(bn);
      rb1 = *(const vfloat4*)(bn + 8);
      if (k0 + 2 * GBK < Kd) {
        __builtin_prefetch(an + GBK, 0, 3);
        __builtin_prefetch(bn + GBK, 0, 3);
      }
    }
    __syncthreads();   // publishes lds[buf]; retires last iter's reads

    const bf16* la = lA[buf];
    const bf16* lb = lB[buf];
    // A fragments (ISA 7.12.2: lanes0-15 K 0-7/16-23, lanes16-31 K 8-15/24-31)
    bf16x16 af[2], bfr[4];
#pragma unroll
    for (int i = 0; i < 2; ++i) {
      const bf16* ar = &la[(wm * 32 + i * 16 + l16) * LAS + half * 8];
      cp16(((vfloat4*)&af[i]) + 0, ar);
      cp16(((vfloat4*)&af[i]) + 1, ar + 16);
    }
    // B fragments: lane = col, K 0-7/16-23 (lanes0-15) / 8-15/24-31 (lanes16-31)
#pragma unroll
    for (int j = 0; j < 4; ++j) {
      const bf16* br = &lb[(wn * 64 + j * 16 + l16) * LBS + half * 8];
      cp16(((vfloat4*)&bfr[j]) + 0, br);
      cp16(((vfloat4*)&bfr[j]) + 1, br + 16);
    }
#pragma unroll
    for (int i = 0; i < 2; ++i)
#pragma unroll
      for (int j = 0; j < 4; ++j)
        acc[i][j] = wmma_bf(af[i], bfr[j], acc[i][j]);

    if (hasnext) {
      bf16* na = lA[buf ^ 1];
      bf16* nb = lB[buf ^ 1];
      *(vfloat4*)&na[sa]     = ra0;
      *(vfloat4*)&na[sa + 8] = ra1;
      *(vfloat4*)&nb[sb]     = rb0;
      *(vfloat4*)&nb[sb + 8] = rb1;
    }
    buf ^= 1;
  }

  // epilogue: C layout = lane->col, vgpr v -> row v + 8*half
#pragma unroll
  for (int i = 0; i < 2; ++i)
#pragma unroll
    for (int j = 0; j < 4; ++j) {
      int col = bn0 + wn * 64 + j * 16 + l16;
      float bv = bias ? bias[col] : 0.f;
#pragma unroll
      for (int v = 0; v < 8; ++v) {
        int row   = bm0 + wm * 32 + i * 16 + v + half * 8;
        float val = acc[i][j][v] + bv;
        if (relu) val = fmaxf(val, 0.f);
        size_t idx = (size_t)row * Nc + col;
        if (outF) __builtin_nontemporal_store(val, &outF[idx]);
        if (outB) outB[idx] = f2bf(val);
      }
    }
}

// ---------------------------------------------------------------------------
// Flash attention: per block = one (b,h) x 64 query rows; 4 waves x 16 rows.
// Q/K in (token, D) layout (head h at cols [h*64,h*64+64)); V pre-transposed
// per (b,h): Vt[((b*H+h)*HD + hd)*S + s].  Double-buffered K/V staging.
// ---------------------------------------------------------------------------
#define AKEYS 32
#define LVS (AKEYS + 8)

__global__ __launch_bounds__(128)
void k_attention(const bf16* __restrict__ Qm, const bf16* __restrict__ Km,
                 const bf16* __restrict__ Vt, bf16* __restrict__ Om, float scale)
{
  __shared__ __align__(16) bf16 lK[2][AKEYS * HD_];     // [key][hd]
  __shared__ __align__(16) bf16 lV[2][HD_ * LVS];       // [hd][key]
  __shared__ __align__(16) bf16 lP[4 * 16 * AKEYS];     // per-wave P tile

  const int tid  = threadIdx.x;
  const int lane = tid & 31;
  const int wave = tid >> 5;
  const int half = lane >> 4;
  const int l16  = lane & 15;
  const int b    = blockIdx.z;
  const int h    = blockIdx.y;
  const int q0   = blockIdx.x * 64 + wave * 16;

  const bf16* Qb = Qm + (size_t)b * S_ * D_ + (size_t)h * HD_;

  // Q fragments for both K-steps of the HD=64 contraction, kept in registers
  bf16x16 qf[2];
#pragma unroll
  for (int s = 0; s < 2; ++s) {
    const bf16* qr = Qb + (size_t)(q0 + l16) * D_ + s * 32 + half * 8;
    cp16(((vfloat4*)&qf[s]) + 0, qr);
    cp16(((vfloat4*)&qf[s]) + 1, qr + 16);
  }

  // staging bases (advance by j0 in the loop)
  const int key = tid >> 2;           // 0..31
  const int kc4 = (tid & 3) * 16;     // 0..48
  const bf16* kbase = Km + (size_t)(b * S_ + key) * D_ + h * HD_ + kc4;
  const int skk = key * HD_ + kc4;
  const int hd  = tid >> 1;           // 0..63
  const int kb2 = (tid & 1) * 16;     // 0 or 16
  const bf16* vbase = Vt + ((size_t)(b * H_ + h) * HD_ + hd) * S_ + kb2;
  const int svv = hd * LVS + kb2;

  // preload key block 0
  vfloat4 rk0 = *(const vfloat4*)(kbase);
  vfloat4 rk1 = *(const vfloat4*)(kbase + 8);
  vfloat4 rv0 = *(const vfloat4*)(vbase);
  vfloat4 rv1 = *(const vfloat4*)(vbase + 8);
  *(vfloat4*)&lK[0][skk]     = rk0;
  *(vfloat4*)&lK[0][skk + 8] = rk1;
  *(vfloat4*)&lV[0][svv]     = rv0;
  *(vfloat4*)&lV[0][svv + 8] = rv1;

  floatx8 o_acc[4] = {};
  float m_run[8], l_run[8];
#pragma unroll
  for (int v = 0; v < 8; ++v) { m_run[v] = -1e30f; l_run[v] = 0.f; }

  int buf = 0;
  for (int j0 = 0; j0 < S_; j0 += AKEYS) {
    const bool hasnext = (j0 + AKEYS) < S_;
    if (hasnext) {
      const bf16* kn = kbase + (size_t)(j0 + AKEYS) * D_;
      const bf16* vn = vbase + (j0 + AKEYS);
      rk0 = *(const vfloat4*)(kn);
      rk1 = *(const vfloat4*)(kn + 8);
      rv0 = *(const vfloat4*)(vn);
      rv1 = *(const vfloat4*)(vn + 8);
    }
    __syncthreads();

    const bf16* lk = lK[buf];
    const bf16* lv = lV[buf];

    // scores tile 16x32 = two 16x16 C tiles, contraction HD=64 (2 k-steps)
    floatx8 sc[2] = {};
#pragma unroll
    for (int s = 0; s < 2; ++s)
#pragma unroll
      for (int t = 0; t < 2; ++t) {
        bf16x16 kf;
        const bf16* kr = &lk[(t * 16 + l16) * HD_ + s * 32 + half * 8];
        cp16(((vfloat4*)&kf) + 0, kr);
        cp16(((vfloat4*)&kf) + 1, kr + 16);
        sc[t] = wmma_bf(qf[s], kf, sc[t]);
      }

    // online softmax: a row lives in one VGPR across a 16-lane half
    float mnew[8], psum[8];
#pragma unroll
    for (int v = 0; v < 8; ++v) {
      float s0 = sc[0][v] * scale, s1 = sc[1][v] * scale;
      float mx = fmaxf(s0, s1);
      for (int m = 1; m < 16; m <<= 1) mx = fmaxf(mx, __shfl_xor(mx, m, 32));
      mnew[v]  = fmaxf(m_run[v], mx);
      float p0 = __expf(s0 - mnew[v]);
      float p1 = __expf(s1 - mnew[v]);
      sc[0][v] = p0; sc[1][v] = p1;
      float ps = p0 + p1;
      for (int m = 1; m < 16; m <<= 1) ps += __shfl_xor(ps, m, 32);
      psum[v] = ps;
    }
#pragma unroll
    for (int v = 0; v < 8; ++v) {
      float alpha = __expf(m_run[v] - mnew[v]);
      l_run[v] = l_run[v] * alpha + psum[v];
      m_run[v] = mnew[v];
#pragma unroll
      for (int n = 0; n < 4; ++n) o_acc[n][v] *= alpha;
    }

    // transpose P (C layout) -> A layout via wave-private LDS tile
    bf16* pw = &lP[wave * 16 * AKEYS];
#pragma unroll
    for (int t = 0; t < 2; ++t)
#pragma unroll
      for (int v = 0; v < 8; ++v)
        pw[(v + half * 8) * AKEYS + t * 16 + l16] = f2bf(sc[t][v]);

    bf16x16 pf;
    {
      const bf16* pr = pw + l16 * AKEYS + half * 8;
      cp16(((vfloat4*)&pf) + 0, pr);
      cp16(((vfloat4*)&pf) + 1, pr + 16);
    }
    // P(16x32) x V(32x64): 4 output tiles over hd
#pragma unroll
    for (int n = 0; n < 4; ++n) {
      bf16x16 vf;
      const bf16* vr = &lv[(n * 16 + l16) * LVS + half * 8];
      cp16(((vfloat4*)&vf) + 0, vr);
      cp16(((vfloat4*)&vf) + 1, vr + 16);
      o_acc[n] = wmma_bf(pf, vf, o_acc[n]);
    }

    if (hasnext) {
      bf16* nk = lK[buf ^ 1];
      bf16* nv = lV[buf ^ 1];
      *(vfloat4*)&nk[skk]     = rk0;
      *(vfloat4*)&nk[skk + 8] = rk1;
      *(vfloat4*)&nv[svv]     = rv0;
      *(vfloat4*)&nv[svv + 8] = rv1;
    }
    buf ^= 1;
  }

  bf16* Ob = Om + (size_t)b * S_ * D_ + (size_t)h * HD_;
#pragma unroll
  for (int n = 0; n < 4; ++n) {
    int col = n * 16 + l16;
#pragma unroll
    for (int v = 0; v < 8; ++v) {
      int row = q0 + v + half * 8;
      Ob[(size_t)row * D_ + col] = f2bf(o_acc[n][v] / l_run[v]);
    }
  }
}

// ---------------------------------------------------------------------------
// LayerNorm over D=1024: out = LN(a (+res)) * sc + bi ; optional bf16 copy
// ---------------------------------------------------------------------------
__global__ __launch_bounds__(256)
void k_layernorm(const float* __restrict__ a, const float* __restrict__ res,
                 const float* __restrict__ sc, const float* __restrict__ bi,
                 float* __restrict__ out, bf16* __restrict__ outbf)
{
  __shared__ float sred[8];
  const size_t row = blockIdx.x;
  const int tid = threadIdx.x;
  const float* ar = a + row * D_;
  const float* rr = res ? res + row * D_ : nullptr;

  float v[4];
  float s = 0.f;
#pragma unroll
  for (int i = 0; i < 4; ++i) {
    int c = tid + i * 256;
    float xv = ar[c];
    if (rr) xv += rr[c];
    v[i] = xv;
    s += xv;
  }
  for (int m = 1; m < 32; m <<= 1) s += __shfl_xor(s, m, 32);
  if ((tid & 31) == 0) sred[tid >> 5] = s;
  __syncthreads();
  float tot = 0.f;
  for (int w = 0; w < 8; ++w) tot += sred[w];
  float mean = tot * (1.f / D_);
  __syncthreads();

  float ss = 0.f;
#pragma unroll
  for (int i = 0; i < 4; ++i) { float d = v[i] - mean; ss += d * d; }
  for (int m = 1; m < 32; m <<= 1) ss += __shfl_xor(ss, m, 32);
  if ((tid & 31) == 0) sred[tid >> 5] = ss;
  __syncthreads();
  float tv = 0.f;
  for (int w = 0; w < 8; ++w) tv += sred[w];
  float rstd = rsqrtf(tv * (1.f / D_) + 1e-5f);

#pragma unroll
  for (int i = 0; i < 4; ++i) {
    int c = tid + i * 256;
    float y = (v[i] - mean) * rstd * sc[c] + bi[c];
    out[row * D_ + c] = y;
    if (outbf) outbf[row * D_ + c] = f2bf(y);
  }
}

// ---------------------------------------------------------------------------
__global__ void k_cvt_bf16(const float* __restrict__ in, bf16* __restrict__ out,
                           size_t n)
{
  size_t i = (size_t)blockIdx.x * blockDim.x + threadIdx.x;
  size_t stride = (size_t)gridDim.x * blockDim.x;
  for (; i < n; i += stride) out[i] = f2bf(in[i]);
}

// fp32 W (K x N) -> bf16 Wt (N x K), LDS-tiled so both sides coalesce
__global__ __launch_bounds__(256)
void k_cvtT(const float* __restrict__ W, bf16* __restrict__ Wt, int K, int Nc)
{
  __shared__ float t[32][33];
  const int n0 = blockIdx.x * 32, k0 = blockIdx.y * 32;
  const int tx = threadIdx.x & 31, ty = threadIdx.x >> 5;
#pragma unroll
  for (int i = 0; i < 32; i += 8)
    t[ty + i][tx] = W[(size_t)(k0 + ty + i) * Nc + n0 + tx];
  __syncthreads();
#pragma unroll
  for (int i = 0; i < 32; i += 8)
    Wt[(size_t)(n0 + ty + i) * K + k0 + tx] = f2bf(t[tx][ty + i]);
}

// V (token,D) bf16 -> Vt[((b*H+h)*HD + hd)*S + s], LDS-tiled transpose
__global__ __launch_bounds__(256)
void k_transpose_v(const bf16* __restrict__ v, bf16* __restrict__ vt)
{
  __shared__ bf16 t[32][33];
  const int bh = blockIdx.z, b = bh / H_, h = bh % H_;
  const int s0 = blockIdx.x * 32, hd0 = blockIdx.y * 32;
  const int tx = threadIdx.x & 31, ty = threadIdx.x >> 5;
#pragma unroll
  for (int i = 0; i < 32; i += 8)
    t[ty + i][tx] = v[(size_t)(b * S_ + s0 + ty + i) * D_ + h * HD_ + hd0 + tx];
  __syncthreads();
#pragma unroll
  for (int i = 0; i < 32; i += 8)
    vt[((size_t)bh * HD_ + hd0 + ty + i) * S_ + s0 + tx] = t[tx][ty + i];
}

// xs = x[:, S-st:, :] -> contiguous fp32 + bf16 copies
__global__ void k_gather_tail(const float* __restrict__ x,
                              float* __restrict__ xf, bf16* __restrict__ xb,
                              int st)
{
  size_t n = (size_t)B_ * st * D_;
  size_t i = (size_t)blockIdx.x * blockDim.x + threadIdx.x;
  size_t stride = (size_t)gridDim.x * blockDim.x;
  for (; i < n; i += stride) {
    size_t r = i / D_;
    int c = (int)(i % D_);
    int b = (int)(r / st);
    int s = S_ - st + (int)(r % st);
    float v = x[((size_t)b * S_ + s) * D_ + c];
    xf[i] = v;
    xb[i] = f2bf(v);
  }
}

// outp = g + pad(y2) + pad(y4) ; also bf16 copy
__global__ void k_combine(const float* __restrict__ g,
                          const float* __restrict__ y2,
                          const float* __restrict__ y4,
                          float* __restrict__ outp, bf16* __restrict__ outb)
{
  size_t n = (size_t)N_ * D_;
  size_t i = (size_t)blockIdx.x * blockDim.x + threadIdx.x;
  size_t stride = (size_t)gridDim.x * blockDim.x;
  for (; i < n; i += stride) {
    size_t t = i / D_;
    int c = (int)(i % D_);
    int b = (int)(t / S_);
    int s = (int)(t % S_);
    float v = g[i];
    if (s >= S_ / 2)
      v += y2[((size_t)b * (S_ / 2) + (s - S_ / 2)) * D_ + c];
    if (s >= S_ - S_ / 4)
      v += y4[((size_t)b * (S_ / 4) + (s - (S_ - S_ / 4))) * D_ + c];
    outp[i] = v;
    outb[i] = f2bf(v);
  }
}

// ---------------------------------------------------------------------------
extern "C" void kernel_launch(void* const* d_in, const int* in_sizes, int n_in,
                              void* d_out, int out_size, void* d_ws, size_t ws_size,
                              hipStream_t stream)
{
  (void)in_sizes; (void)n_in; (void)out_size; (void)ws_size;

  const float* x    = (const float*)d_in[0];
  const float* Wq   = (const float*)d_in[1];
  const float* bq   = (const float*)d_in[2];
  const float* Wk   = (const float*)d_in[3];
  const float* bk   = (const float*)d_in[4];
  const float* Wv   = (const float*)d_in[5];
  const float* bv   = (const float*)d_in[6];
  const float* Wo   = (const float*)d_in[7];
  const float* bo   = (const float*)d_in[8];
  const float* W1   = (const float*)d_in[9];
  const float* b1   = (const float*)d_in[10];
  const float* W2   = (const float*)d_in[11];
  const float* b2   = (const float*)d_in[12];
  const float* n1s  = (const float*)d_in[13];
  const float* n1b  = (const float*)d_in[14];
  const float* n2s  = (const float*)d_in[15];
  const float* n2b  = (const float*)d_in[16];
  const float* ln1s = (const float*)d_in[17];
  const float* ln1b = (const float*)d_in[18];
  const float* ln2s = (const float*)d_in[19];
  const float* ln2b = (const float*)d_in[20];
  const float* ln3s = (const float*)d_in[21];
  const float* ln3b = (const float*)d_in[22];
  const float* Wb   = (const float*)d_in[23];
  const float* bb   = (const float*)d_in[24];
  const float* Wm   = (const float*)d_in[25];
  const float* bm   = (const float*)d_in[26];
  const float* Wf1  = (const float*)d_in[27];
  const float* bf1  = (const float*)d_in[28];
  const float* Wf2  = (const float*)d_in[29];
  const float* bf2  = (const float*)d_in[30];

  char* ws = (char*)d_ws;
  size_t off = 0;
  auto alloc = [&](size_t bytes) -> char* {
    char* p = ws + off;
    off += (bytes + 255) & ~(size_t)255;
    return p;
  };

  // bf16 TRANSPOSED weight pool (N x K layouts), rebuilt every launch
  bf16* wq_bf  = (bf16*)alloc((size_t)L_ * D_ * D_ * 2);
  bf16* wk_bf  = (bf16*)alloc((size_t)L_ * D_ * D_ * 2);
  bf16* wv_bf  = (bf16*)alloc((size_t)L_ * D_ * D_ * 2);
  bf16* wo_bf  = (bf16*)alloc((size_t)L_ * D_ * D_ * 2);
  bf16* w1_bf  = (bf16*)alloc((size_t)L_ * D_ * FD_ * 2);
  bf16* w2_bf  = (bf16*)alloc((size_t)L_ * FD_ * D_ * 2);
  bf16* wb_bf  = (bf16*)alloc((size_t)D_ * MM_ * 2);
  bf16* wm_bf  = (bf16*)alloc((size_t)MM_ * D_ * 2);
  bf16* wf1_bf = (bf16*)alloc((size_t)D_ * FD_ * 2);
  bf16* wf2_bf = (bf16*)alloc((size_t)FD_ * D_ * 2);

  float* g     = (float*)alloc((size_t)N_ * D_ * 4);
  float* tmpD  = (float*)alloc((size_t)N_ * D_ * 4);
  bf16* a_bf   = (bf16*)alloc((size_t)N_ * D_ * 2);
  bf16* hid_bf = (bf16*)alloc((size_t)N_ * FD_ * 2);
  bf16* q_bf   = (bf16*)alloc((size_t)N_ * D_ * 2);
  bf16* kk_bf  = (bf16*)alloc((size_t)N_ * D_ * 2);
  bf16* v_bf   = (bf16*)alloc((size_t)N_ * D_ * 2);
  bf16* vt_bf  = (bf16*)alloc((size_t)N_ * D_ * 2);
  bf16* ao_bf  = (bf16*)alloc((size_t)N_ * D_ * 2);

  const int R2 = B_ * (S_ / 2), R4 = B_ * (S_ / 4);
  float* xs2_f  = (float*)alloc((size_t)R2 * D_ * 4);
  bf16*  xs2_bf = (bf16*)alloc((size_t)R2 * D_ * 2);
  bf16*  u2_bf  = (bf16*)alloc((size_t)R2 * MM_ * 2);
  float* mo2    = (float*)alloc((size_t)R2 * D_ * 4);
  float* y2     = (float*)alloc((size_t)R2 * D_ * 4);
  float* xs4_f  = (float*)alloc((size_t)R4 * D_ * 4);
  bf16*  xs4_bf = (bf16*)alloc((size_t)R4 * D_ * 2);
  bf16*  u4_bf  = (bf16*)alloc((size_t)R4 * MM_ * 2);
  float* mo4    = (float*)alloc((size_t)R4 * D_ * 4);
  float* y4     = (float*)alloc((size_t)R4 * D_ * 4);
  float* outp   = (float*)alloc((size_t)N_ * D_ * 4);
  bf16*  outp_bf= (bf16*)alloc((size_t)N_ * D_ * 2);
  float* hbuf   = (float*)alloc((size_t)N_ * D_ * 4);

  auto cvt = [&](const float* in, bf16* out, size_t n) {
    k_cvt_bf16<<<4096, 256, 0, stream>>>(in, out, n);
  };
  auto cvtT = [&](const float* W, bf16* Wt, int K, int Nc) {
    k_cvtT<<<dim3(Nc / 32, K / 32), 256, 0, stream>>>(W, Wt, K, Nc);
  };
  auto gemm = [&](const bf16* A, const bf16* Bt, const float* bias,
                  float* oF, bf16* oB, int Mr, int Nc, int Kd, int relu) {
    dim3 gr(Nc / GBN, Mr / GBM);
    k_gemm_bf16<<<gr, 256, 0, stream>>>(A, Bt, bias, oF, oB, Mr, Nc, Kd, relu);
  };
  auto ln = [&](int rows, const float* a, const float* res, const float* sc,
                const float* bi, float* out, bf16* obf) {
    k_layernorm<<<rows, 256, 0, stream>>>(a, res, sc, bi, out, obf);
  };

  // convert + transpose weights
  for (int l = 0; l < L_; ++l) {
    const size_t dd = (size_t)l * D_ * D_;
    const size_t df = (size_t)l * D_ * FD_;
    cvtT(Wq + dd, wq_bf + dd, D_, D_);
    cvtT(Wk + dd, wk_bf + dd, D_, D_);
    cvtT(Wv + dd, wv_bf + dd, D_, D_);
    cvtT(Wo + dd, wo_bf + dd, D_, D_);
    cvtT(W1 + df, w1_bf + df, D_, FD_);
    cvtT(W2 + df, w2_bf + df, FD_, D_);
  }
  cvtT(Wb, wb_bf, D_, MM_);
  cvtT(Wm, wm_bf, MM_, D_);
  cvtT(Wf1, wf1_bf, D_, FD_);
  cvtT(Wf2, wf2_bf, FD_, D_);

  // g = x; a_bf = bf16(x)
  hipMemcpyAsync(g, x, (size_t)N_ * D_ * 4, hipMemcpyDeviceToDevice, stream);
  cvt(x, a_bf, (size_t)N_ * D_);

  for (int i = 0; i < L_; ++i) {
    const size_t dd = (size_t)i * D_ * D_;
    const size_t df = (size_t)i * D_ * FD_;
    gemm(a_bf, wq_bf + dd, bq + i * D_, nullptr, q_bf, N_, D_, D_, 0);
    gemm(a_bf, wk_bf + dd, bk + i * D_, nullptr, kk_bf, N_, D_, D_, 0);
    gemm(a_bf, wv_bf + dd, bv + i * D_, nullptr, v_bf, N_, D_, D_, 0);
    k_transpose_v<<<dim3(S_ / 32, HD_ / 32, B_ * H_), 256, 0, stream>>>(v_bf,
                                                                        vt_bf);
    k_attention<<<dim3(S_ / 64, H_, B_), 128, 0, stream>>>(q_bf, kk_bf, vt_bf,
                                                           ao_bf, 0.125f);
    gemm(ao_bf, wo_bf + dd, bo + i * D_, tmpD, nullptr, N_, D_, D_, 0);
    ln(N_, tmpD, g, n1s + i * D_, n1b + i * D_, g, a_bf);
    gemm(a_bf, w1_bf + df, b1 + i * FD_, nullptr, hid_bf, N_, FD_, D_, 1);
    gemm(hid_bf, w2_bf + df, b2 + i * D_, tmpD, nullptr, N_, D_, FD_, 0);
    ln(N_, tmpD, g, n2s + i * D_, n2b + i * D_, g, a_bf);
  }

  // local(2)
  k_gather_tail<<<2048, 256, 0, stream>>>(x, xs2_f, xs2_bf, S_ / 2);
  gemm(xs2_bf, wb_bf, bb, nullptr, u2_bf, R2, MM_, D_, 0);
  gemm(u2_bf, wm_bf, bm, mo2, nullptr, R2, D_, MM_, 0);
  ln(R2, mo2, xs2_f, ln1s, ln1b, y2, nullptr);

  // local(4)
  k_gather_tail<<<2048, 256, 0, stream>>>(x, xs4_f, xs4_bf, S_ / 4);
  gemm(xs4_bf, wb_bf, bb, nullptr, u4_bf, R4, MM_, D_, 0);
  gemm(u4_bf, wm_bf, bm, mo4, nullptr, R4, D_, MM_, 0);
  ln(R4, mo4, xs4_f, ln1s, ln1b, y4, nullptr);

  // output = g + pad(y2) + pad(y4)
  k_combine<<<4096, 256, 0, stream>>>(g, y2, y4, outp, outp_bf);
  // h = LN(output + x)
  ln(N_, outp, x, ln2s, ln2b, hbuf, nullptr);
  // ff = relu(output@Wf1+bf1)@Wf2+bf2
  gemm(outp_bf, wf1_bf, bf1, nullptr, hid_bf, N_, FD_, D_, 1);
  gemm(hid_bf, wf2_bf, bf2, tmpD, nullptr, N_, D_, FD_, 0);
  // out = LN(h + ff)
  ln(N_, tmpD, hbuf, ln3s, ln3b, (float*)d_out, nullptr);
}
